// AttentionHead_12446815224315
// MI455X (gfx1250) — matrix operand order repaired and checked by reading, not compiled, not run on hardware
//
#include <hip/hip_runtime.h>

#define H_DIM 128

typedef __attribute__((ext_vector_type(16))) __bf16    v16bf;
typedef __attribute__((ext_vector_type(8)))  float     v8f;
typedef __attribute__((ext_vector_type(8)))  unsigned  v8u;

__device__ __forceinline__ unsigned short f2bf(float f) {
    unsigned u = __float_as_uint(f);
    u += 0x7FFFu + ((u >> 16) & 1u);          // round-to-nearest-even
    return (unsigned short)(u >> 16);
}

// ---------------------------------------------------------------------------
// Kernel 1: fused QKV projection, fp32 -> bf16, WMMA bf16 GEMM.
// Block = 256 threads (8 waves). Each block: 128 rows x 128 cols (all of H).
// blockIdx.y selects Wq / Wk / Wv; output planes are contiguous in d_ws.
// ---------------------------------------------------------------------------
__global__ __launch_bounds__(256) void qkv_proj_kernel(
    const float* __restrict__ x,
    const float* __restrict__ Wq,
    const float* __restrict__ Wk,
    const float* __restrict__ Wv,
    unsigned short* __restrict__ qkv, // 3 planes of N*H bf16
    int Nrows, int E)
{
    __shared__ unsigned short lx[128 * 32];  // x tile, row-major [row][k]
    __shared__ unsigned short lw[128 * 32];  // W tile, transposed [n][k]

    const int t    = threadIdx.x;
    const int wave = t >> 5;
    const int lane = t & 31;
    const int lrow = lane & 15;
    const int hi   = lane >> 4;
    const int row0 = blockIdx.x * 128;
    const int which = blockIdx.y;

    const float* W = (which == 0) ? Wq : (which == 1) ? Wk : Wv;
    unsigned short* outp = qkv + (size_t)which * (size_t)Nrows * H_DIM;

    v8f acc[8];
#pragma unroll
    for (int i = 0; i < 8; ++i) acc[i] = (v8f){0,0,0,0,0,0,0,0};

    for (int kk = 0; kk < E; kk += 32) {
        // stage x tile (128x32) as bf16, row-major
#pragma unroll
        for (int i = 0; i < 16; ++i) {
            int idx = t + i * 256;
            int r = idx >> 5, k = idx & 31;
            lx[r * 32 + k] = f2bf(x[(size_t)(row0 + r) * E + kk + k]);
        }
        // stage W tile (32x128) as bf16, transposed -> [n][k] so K-pairs are contiguous
#pragma unroll
        for (int i = 0; i < 16; ++i) {
            int idx = t + i * 256;
            int k = idx >> 7, n = idx & 127;
            lw[n * 32 + k] = f2bf(W[(size_t)(kk + k) * H_DIM + n]);
        }
        __syncthreads();

        // A fragment (16x32 bf16): lane holds row (lane&15), K-half by (lane>>4)
        v8u au;
#pragma unroll
        for (int v = 0; v < 4; ++v)
            au[v] = *(const unsigned*)&lx[(wave * 16 + lrow) * 32 + hi * 8 + 2 * v];
#pragma unroll
        for (int v = 0; v < 4; ++v)
            au[4 + v] = *(const unsigned*)&lx[(wave * 16 + lrow) * 32 + hi * 8 + 16 + 2 * v];
        v16bf a = __builtin_bit_cast(v16bf, au);

#pragma unroll
        for (int tc = 0; tc < 8; ++tc) {
            // B fragment (32x16): lane holds column n, VGPR v holds K = (lane>>4)*8 + 2(v&3) + (v>>2)*16
            v8u bu;
#pragma unroll
            for (int v = 0; v < 8; ++v)
                bu[v] = *(const unsigned*)&lw[(tc * 16 + lrow) * 32 + hi * 8 + 2 * (v & 3) + (v >> 2) * 16];
            v16bf bfr = __builtin_bit_cast(v16bf, bu);
            acc[tc] = __builtin_amdgcn_wmma_f32_16x16x32_bf16(
                false, a, false, bfr, (short)0, acc[tc], false, false);
        }
        __syncthreads();
    }

    // C layout: VGPR r -> row r (+8 for upper half-wave), lane&15 -> column
#pragma unroll
    for (int tc = 0; tc < 8; ++tc)
#pragma unroll
        for (int r = 0; r < 8; ++r) {
            int row = row0 + wave * 16 + r + 8 * hi;
            outp[(size_t)row * H_DIM + tc * 16 + lrow] = f2bf(acc[tc][r]);
        }
}

// ---------------------------------------------------------------------------
// Kernel 2: causal flash attention over bf16 Q/K/V, f32 accumulation.
// Block = 256 threads (8 waves) handles 128 query rows of one batch.
// Keys streamed in 64-key LDS blocks. Online softmax, lane-shuffle row stats.
// ---------------------------------------------------------------------------
__global__ __launch_bounds__(256) void flash_attn_kernel(
    const unsigned short* __restrict__ Qbf,
    const unsigned short* __restrict__ Kbf,
    const unsigned short* __restrict__ Vbf,
    const int*            __restrict__ k_mask,
    float*                __restrict__ out,
    int Sdim)
{
    __shared__ unsigned short lk[64 * 128];       // K tile [key][h]      16 KB
    __shared__ unsigned short lv[128 * 64];       // V tile transposed [h][key] 16 KB
    __shared__ unsigned short lp[8][16 * 64];     // per-wave P tile      16 KB

    const float SCALE = 0.08838834764831845f;     // 1/sqrt(128)
    const int t    = threadIdx.x;
    const int wave = t >> 5;
    const int lane = t & 31;
    const int lrow = lane & 15;
    const int hi   = lane >> 4;
    const int b    = blockIdx.y;
    const int qblk = blockIdx.x;
    const int q0   = qblk * 128;
    const size_t base = (size_t)b * Sdim;

    // Resident Q A-fragments: 4 K-chunks of 32 over H. Pairs along H are
    // contiguous in global memory, matching the ISA A-layout directly.
    v16bf qa[4];
    {
        const unsigned short* qp = Qbf + (base + q0 + wave * 16 + lrow) * H_DIM;
#pragma unroll
        for (int c = 0; c < 4; ++c) {
            v8u u;
#pragma unroll
            for (int v = 0; v < 4; ++v)
                u[v] = *(const unsigned*)&qp[c * 32 + hi * 8 + 2 * v];
#pragma unroll
            for (int v = 0; v < 4; ++v)
                u[4 + v] = *(const unsigned*)&qp[c * 32 + hi * 8 + 16 + 2 * v];
            qa[c] = __builtin_bit_cast(v16bf, u);
        }
    }

    v8f O[8];
    float m[8], lsum[8];
#pragma unroll
    for (int i = 0; i < 8; ++i) { O[i] = (v8f){0,0,0,0,0,0,0,0}; m[i] = -3.0e38f; lsum[i] = 0.0f; }

    const int jmax = 2 * qblk + 1;                // last 64-key block touching q0+127
    for (int j = 0; j <= jmax; ++j) {
        if (j < jmax)                             // global_prefetch_b8 of next K block
            __builtin_prefetch(Kbf + (base + (size_t)(j + 1) * 64) * H_DIM, 0, 1);

        // stage K tile (64 keys x 128 h), row-major, dword copies
        const unsigned* kg = (const unsigned*)(Kbf + (base + (size_t)j * 64) * H_DIM);
        unsigned* lku = (unsigned*)lk;
#pragma unroll
        for (int i = 0; i < 16; ++i) lku[t + i * 256] = kg[t + i * 256];
        // stage V tile transposed -> [h][key]
        const unsigned short* vg = Vbf + (base + (size_t)j * 64) * H_DIM;
#pragma unroll
        for (int i = 0; i < 32; ++i) {
            int di = t + i * 256;
            int key = di >> 7, h = di & 127;
            lv[h * 64 + key] = vg[key * H_DIM + h];
        }
        __syncthreads();

        // S = Q K^T : 4 key-tiles x 4 h-chunks of WMMA
        v8f sacc[4];
#pragma unroll
        for (int i = 0; i < 4; ++i) sacc[i] = (v8f){0,0,0,0,0,0,0,0};
#pragma unroll
        for (int tt = 0; tt < 4; ++tt)
#pragma unroll
            for (int c = 0; c < 4; ++c) {
                v8u bu;
#pragma unroll
                for (int v = 0; v < 8; ++v)
                    bu[v] = *(const unsigned*)&lk[(tt * 16 + lrow) * 128 + c * 32 + hi * 8 + 2 * (v & 3) + (v >> 2) * 16];
                v16bf bk = __builtin_bit_cast(v16bf, bu);
                sacc[tt] = __builtin_amdgcn_wmma_f32_16x16x32_bf16(
                    false, qa[c], false, bk, (short)0, sacc[tt], false, false);
            }

        // scale + causal + key-padding mask, running row max
        float mnew[8];
#pragma unroll
        for (int r = 0; r < 8; ++r) mnew[r] = -3.0e38f;
#pragma unroll
        for (int tt = 0; tt < 4; ++tt) {
            int keyg = j * 64 + tt * 16 + lrow;
            float kmadd = (k_mask[base + keyg] == 0) ? -3.0e38f : 0.0f;
#pragma unroll
            for (int r = 0; r < 8; ++r) {
                int rowg = q0 + wave * 16 + r + 8 * hi;
                float sv = sacc[tt][r] * SCALE + kmadd;
                if (keyg > rowg) sv = -3.0e38f;   // causal
                sacc[tt][r] = sv;
                mnew[r] = fmaxf(mnew[r], sv);
            }
        }
        // reduce max across the 16-lane half that shares a matrix row
#pragma unroll
        for (int r = 0; r < 8; ++r) {
#pragma unroll
            for (int off = 8; off > 0; off >>= 1)
                mnew[r] = fmaxf(mnew[r], __shfl_xor(mnew[r], off, 32));
            float mc    = fmaxf(m[r], mnew[r]);
            float alpha = __expf(m[r] - mc);
            m[r] = mc;
            lsum[r] *= alpha;
#pragma unroll
            for (int tt = 0; tt < 8; ++tt) O[tt][r] *= alpha;
        }
        // p = exp(s - m): row sums + bf16 P tile into per-wave LDS (A-layout source)
#pragma unroll
        for (int r = 0; r < 8; ++r) {
            float ps = 0.0f;
#pragma unroll
            for (int tt = 0; tt < 4; ++tt) {
                float p = __expf(sacc[tt][r] - m[r]);
                ps += p;
                lp[wave][(r + 8 * hi) * 64 + tt * 16 + lrow] = f2bf(p);
            }
#pragma unroll
            for (int off = 8; off > 0; off >>= 1)
                ps += __shfl_xor(ps, off, 32);
            lsum[r] += ps;
        }
        asm volatile("s_wait_dscnt 0" ::: "memory"); // P store -> P fragment reload (same wave)

        // O += P V : 2 key-chunks x 8 h-tiles of WMMA
#pragma unroll
        for (int c = 0; c < 2; ++c) {
            v8u au;
#pragma unroll
            for (int v = 0; v < 4; ++v)
                au[v] = *(const unsigned*)&lp[wave][lrow * 64 + c * 32 + hi * 8 + 2 * v];
#pragma unroll
            for (int v = 0; v < 4; ++v)
                au[4 + v] = *(const unsigned*)&lp[wave][lrow * 64 + c * 32 + hi * 8 + 16 + 2 * v];
            v16bf pa = __builtin_bit_cast(v16bf, au);
#pragma unroll
            for (int tt = 0; tt < 8; ++tt) {
                v8u bu;
#pragma unroll
                for (int v = 0; v < 8; ++v)
                    bu[v] = *(const unsigned*)&lv[(tt * 16 + lrow) * 64 + c * 32 + hi * 8 + 2 * (v & 3) + (v >> 2) * 16];
                v16bf bv = __builtin_bit_cast(v16bf, bu);
                O[tt] = __builtin_amdgcn_wmma_f32_16x16x32_bf16(
                    false, pa, false, bv, (short)0, O[tt], false, false);
            }
        }
        __syncthreads();
    }

    // epilogue: O / rowsum, fp32 out
#pragma unroll
    for (int r = 0; r < 8; ++r) {
        float rl = 1.0f / lsum[r];
        int rowg = q0 + wave * 16 + r + 8 * hi;
#pragma unroll
        for (int tt = 0; tt < 8; ++tt)
            out[(base + rowg) * H_DIM + tt * 16 + lrow] = O[tt][r] * rl;
    }
}

// ---------------------------------------------------------------------------
extern "C" void kernel_launch(void* const* d_in, const int* in_sizes, int n_in,
                              void* d_out, int out_size, void* d_ws, size_t ws_size,
                              hipStream_t stream) {
    (void)n_in; (void)out_size; (void)ws_size;
    const float* x  = (const float*)d_in[0];
    const float* Wq = (const float*)d_in[1];
    const float* Wk = (const float*)d_in[2];
    const float* Wv = (const float*)d_in[3];
    const int* km   = (const int*)d_in[4];
    float* out      = (float*)d_out;

    const int E = in_sizes[1] / H_DIM;      // 1024
    const int N = in_sizes[0] / E;          // B*S = 16384
    const int B = 4;
    const int S = N / B;                    // 4096

    unsigned short* Qbf = (unsigned short*)d_ws;  // 3 contiguous bf16 planes of N*H
    unsigned short* Kbf = Qbf + (size_t)N * H_DIM;
    unsigned short* Vbf = Kbf + (size_t)N * H_DIM;

    dim3 gA(N / 128, 3);
    qkv_proj_kernel<<<gA, 256, 0, stream>>>(x, Wq, Wk, Wv, Qbf, N, E);

    dim3 gB(S / 128, B);
    flash_attn_kernel<<<gB, 256, 0, stream>>>(Qbf, Kbf, Vbf, km, out, S);
}
